// MultiHead_33200097198987
// MI455X (gfx1250) — matrix operand order repaired and checked by reading, not compiled, not run on hardware
//
#include <hip/hip_runtime.h>
#include <hip/hip_bf16.h>
#include <cstdint>

#define EMBED 1024
#define HEADS 16
#define DK    64
#define SEQ   2048
#define BATCH 2

typedef __bf16 bf16;
typedef __attribute__((ext_vector_type(16))) __bf16 v16bf;
typedef __attribute__((ext_vector_type(8)))  __bf16 v8bf;
typedef __attribute__((ext_vector_type(8)))  float  v8f;
typedef __attribute__((ext_vector_type(4)))  float  v4f;
typedef __attribute__((ext_vector_type(4)))  int    v4i;

// ---------------------------------------------------------------------------
// gfx1250 async global->LDS copy (16B per lane, tracked on ASYNCcnt).
// Builtin signature (from hipcc diagnostic): (v4i AS1*, v4i AS3*, imm, imm).
// Guarded: falls back to a plain VGPR round-trip copy if the builtin is
// absent in this toolchain.
// ---------------------------------------------------------------------------
#if defined(__has_builtin)
#if __has_builtin(__builtin_amdgcn_global_load_async_to_lds_b128)
#define HAVE_ASYNC_COPY 1
#endif
#endif

#ifdef HAVE_ASYNC_COPY
typedef __attribute__((address_space(1))) v4i gv4i;   // global (AS1)
typedef __attribute__((address_space(3))) v4i lv4i;   // LDS (AS3)
__device__ inline void async_copy16(const void* gsrc, void* ldst) {
  gv4i* g = (gv4i*)(uintptr_t)gsrc;
  // Low 32 bits of a generic LDS pointer are the LDS byte offset
  lv4i* l = (lv4i*)(uint32_t)(uintptr_t)ldst;
  __builtin_amdgcn_global_load_async_to_lds_b128(g, l, 0, 0);
}
__device__ inline void wait_async0() {
  asm volatile("s_wait_asynccnt 0" ::: "memory");
}
#else
__device__ inline void async_copy16(const void* gsrc, void* ldst) {
  *(v8bf*)ldst = *(const v8bf*)gsrc;
}
__device__ inline void wait_async0() {}
#endif

// Build a 16-element bf16 WMMA fragment from two contiguous 8-element runs.
// (CDNA5 16-bit A/B fragments are two 8-halfword runs per lane.)
__device__ inline v16bf load_frag_pair(const bf16* p0, const bf16* p1) {
  v8bf lo = *(const v8bf*)p0;
  v8bf hi = *(const v8bf*)p1;
  v16bf r;
#pragma unroll
  for (int i = 0; i < 8; ++i) { r[i] = lo[i]; r[i + 8] = hi[i]; }
  return r;
}

// ---------------------------------------------------------------------------
// Pass 1: Out[b,h,s,d] = bf16( X[m,:] @ W[:,n] + bias[n] ),  m=b*S+s, n=h*64+d
// X: f32 [4096 x 1024], W: f32 [1024 x 1024]. Block tile 128x128, 8 waves
// (4 along M x 2 along N), each wave owns 32x64 (2x4 WMMA tiles). K step 32.
// ---------------------------------------------------------------------------
__global__ __launch_bounds__(256)
void proj_kernel(const float* __restrict__ X, const float* __restrict__ W,
                 const float* __restrict__ bias, bf16* __restrict__ Out) {
  __shared__ bf16 As[128][32];   // A tile, row-major (M x K)
  __shared__ bf16 Wt[128][32];   // W tile, transposed (N x K)

  const int tid  = threadIdx.x;
  const int lane = tid & 31;
  const int wave = tid >> 5;
  const int hi   = lane >> 4;
  const int l15  = lane & 15;
  const int mBase = blockIdx.x * 128;
  const int nBase = blockIdx.y * 128;
  const int wm = wave & 3;   // 4 waves along M (32 rows each)
  const int wn = wave >> 2;  // 2 waves along N (64 cols each)

  v8f acc[2][4] = {};

  for (int kt = 0; kt < EMBED / 32; ++kt) {
    // --- stage A: 128x32 f32 -> bf16 (16 elements per thread) ---
    {
      const int r  = tid >> 1;
      const int c0 = (tid & 1) * 16;
      const float* src = X + (size_t)(mBase + r) * EMBED + kt * 32 + c0;
      if (kt + 1 < EMBED / 32) __builtin_prefetch(src + 32, 0, 1);
#pragma unroll
      for (int i = 0; i < 16; i += 4) {
        v4f v = *(const v4f*)(src + i);
        As[r][c0 + i + 0] = (bf16)v[0];
        As[r][c0 + i + 1] = (bf16)v[1];
        As[r][c0 + i + 2] = (bf16)v[2];
        As[r][c0 + i + 3] = (bf16)v[3];
      }
    }
    // --- stage W transposed: 32x128 f32 -> Wt[n][k] bf16 (16 per thread) ---
    {
      const int k  = tid & 31;
      const int n0 = (tid >> 5) * 16;
      const float* src = W + (size_t)(kt * 32 + k) * EMBED + nBase + n0;
      if (kt + 1 < EMBED / 32) __builtin_prefetch(src + 32 * EMBED, 0, 1);
#pragma unroll
      for (int i = 0; i < 16; i += 4) {
        v4f v = *(const v4f*)(src + i);
        Wt[n0 + i + 0][k] = (bf16)v[0];
        Wt[n0 + i + 1][k] = (bf16)v[1];
        Wt[n0 + i + 2][k] = (bf16)v[2];
        Wt[n0 + i + 3][k] = (bf16)v[3];
      }
    }
    __syncthreads();

    v16bf a[2], b[4];
#pragma unroll
    for (int mt = 0; mt < 2; ++mt) {
      const bf16* row = &As[wm * 32 + mt * 16 + l15][0];
      a[mt] = load_frag_pair(row + hi * 8, row + 16 + hi * 8);
    }
#pragma unroll
    for (int nt = 0; nt < 4; ++nt) {
      const bf16* col = &Wt[wn * 64 + nt * 16 + l15][0];
      b[nt] = load_frag_pair(col + hi * 8, col + 16 + hi * 8);
    }
#pragma unroll
    for (int mt = 0; mt < 2; ++mt)
#pragma unroll
      for (int nt = 0; nt < 4; ++nt)
        acc[mt][nt] = __builtin_amdgcn_wmma_f32_16x16x32_bf16(
            false, a[mt], false, b[nt], (short)0, acc[mt][nt], false, false);
    __syncthreads();
  }

  // epilogue: +bias, convert to bf16, scatter into [B,H,S,DK] head layout
#pragma unroll
  for (int mt = 0; mt < 2; ++mt) {
#pragma unroll
    for (int nt = 0; nt < 4; ++nt) {
      const int ncol = nBase + wn * 64 + nt * 16 + l15;
      const int h = ncol >> 6, d = ncol & 63;
      const float bv = bias[ncol];
#pragma unroll
      for (int r = 0; r < 8; ++r) {
        const int mrow = mBase + wm * 32 + mt * 16 + r + hi * 8;
        const int bidx = mrow >> 11;          // / SEQ
        const int s    = mrow & (SEQ - 1);
        const float val = acc[mt][nt][r] + bv;
        Out[(((size_t)bidx * HEADS + h) * SEQ + s) * DK + d] = (bf16)val;
      }
    }
  }
}

// ---------------------------------------------------------------------------
// Pass 2: flash attention per (b,h). Block = 8 waves x 16 query rows = 128
// rows. Keys/values streamed in 32-key chunks through LDS; online softmax.
// K chunk is staged with GLOBAL_LOAD_ASYNC_TO_LDS_B128 (pure bf16 copy).
// Reference quirk preserved: masked scores become 1e-8 *before* softmax.
// Ctx written bf16 in [B,S,H,DK] (row-major [4096 x 1024]) for the out proj.
// ---------------------------------------------------------------------------
__global__ __launch_bounds__(256)
void attn_kernel(const bf16* __restrict__ Q, const bf16* __restrict__ K,
                 const bf16* __restrict__ V, const int* __restrict__ mask,
                 bf16* __restrict__ Ctx) {
  __shared__ bf16 Kc[32][64];       // K chunk, row-major [key][d]
  __shared__ bf16 Vt[64][32];       // V chunk, transposed [d][key]
  __shared__ bf16 Pb[8][16][32];    // per-wave P tile (C-layout -> A-layout)

  const int tid  = threadIdx.x;
  const int lane = tid & 31;
  const int wave = tid >> 5;
  const int hi   = lane >> 4;
  const int l15  = lane & 15;
  const int bh   = blockIdx.y;          // 0..31
  const int b    = bh >> 4, h = bh & 15;
  const int qBase = blockIdx.x * 128 + wave * 16;

  const bf16* Qh = Q + (size_t)bh * SEQ * DK;
  const bf16* Kh = K + (size_t)bh * SEQ * DK;
  const bf16* Vh = V + (size_t)bh * SEQ * DK;
  const int*  Mb = mask + (size_t)b * SEQ * SEQ;

  // Q fragments for this wave's 16 rows: Dk=64 -> 2 K-tiles of 32
  v16bf qf[2];
#pragma unroll
  for (int t = 0; t < 2; ++t) {
    const bf16* row = Qh + (size_t)(qBase + l15) * DK + t * 32;
    qf[t] = load_frag_pair(row + hi * 8, row + 16 + hi * 8);
  }

  v8f ctx[4] = {};
  float mrow[8], lrow[8];
#pragma unroll
  for (int r = 0; r < 8; ++r) { mrow[r] = -1e30f; lrow[r] = 0.0f; }

  for (int kc = 0; kc < SEQ / 32; ++kc) {
    const int keyBase = kc * 32;
    // --- stage K chunk asynchronously (one b128 per lane), V transposed ---
    {
      const int kk = tid >> 3;         // 0..31 key in chunk
      const int d0 = (tid & 7) * 8;    // 0..56
      async_copy16(Kh + (size_t)(keyBase + kk) * DK + d0, &Kc[kk][d0]);
      const bf16* sv = Vh + (size_t)(keyBase + kk) * DK + d0;
      if (kc + 1 < SEQ / 32) __builtin_prefetch(sv + 32 * DK, 0, 1);
      v8bf vv = *(const v8bf*)sv;
#pragma unroll
      for (int i = 0; i < 8; ++i) Vt[d0 + i][kk] = vv[i];
    }
    wait_async0();
    __syncthreads();

    // --- scores: Q[16x64] x K_chunk^T -> 16x32 (2 N-tiles, 2 K-steps) ---
    v8f sc[2] = {};
#pragma unroll
    for (int nt = 0; nt < 2; ++nt) {
#pragma unroll
      for (int t = 0; t < 2; ++t) {
        const bf16* col = &Kc[nt * 16 + l15][t * 32 + hi * 8];
        v16bf bf = load_frag_pair(col, col + 16);
        sc[nt] = __builtin_amdgcn_wmma_f32_16x16x32_bf16(
            false, qf[t], false, bf, (short)0, sc[nt], false, false);
      }
    }

    // --- scale 1/sqrt(64) + reference masking (1e-8, not -inf) ---
#pragma unroll
    for (int nt = 0; nt < 2; ++nt) {
      const int key = keyBase + nt * 16 + l15;   // C layout: col = lane&15
      const int* mp = Mb + key;
#pragma unroll
      for (int r = 0; r < 8; ++r) {
        const int qrow = qBase + r + hi * 8;     // C layout: row = r + 8*hi
        const int mv = mp[(size_t)qrow * SEQ];
        const float s = sc[nt][r] * 0.125f;
        sc[nt][r] = (mv != 0) ? s : 1e-8f;
      }
    }

    // --- online softmax (row reductions across the 16-lane row groups) ---
#pragma unroll
    for (int r = 0; r < 8; ++r) {
      float v = fmaxf(sc[0][r], sc[1][r]);
#pragma unroll
      for (int off = 1; off < 16; off <<= 1)
        v = fmaxf(v, __shfl_xor(v, off, 32));
      const float newm  = fmaxf(mrow[r], v);
      const float scale = __expf(mrow[r] - newm);
      const float p0 = __expf(sc[0][r] - newm);
      const float p1 = __expf(sc[1][r] - newm);
      float psum = p0 + p1;
#pragma unroll
      for (int off = 1; off < 16; off <<= 1)
        psum += __shfl_xor(psum, off, 32);
      lrow[r] = lrow[r] * scale + psum;
      mrow[r] = newm;
#pragma unroll
      for (int nt = 0; nt < 4; ++nt) ctx[nt][r] *= scale;
      Pb[wave][r + hi * 8][l15]      = (bf16)p0;   // C-layout -> LDS
      Pb[wave][r + hi * 8][16 + l15] = (bf16)p1;
    }

    // --- P (A-layout) x V_chunk -> ctx accumulate (4 N-tiles over Dk) ---
    const bf16* prow = &Pb[wave][l15][0];
    v16bf pf = load_frag_pair(prow + hi * 8, prow + 16 + hi * 8);
#pragma unroll
    for (int nt = 0; nt < 4; ++nt) {
      const bf16* col = &Vt[nt * 16 + l15][hi * 8];
      v16bf bf = load_frag_pair(col, col + 16);
      ctx[nt] = __builtin_amdgcn_wmma_f32_16x16x32_bf16(
          false, pf, false, bf, (short)0, ctx[nt], false, false);
    }
    __syncthreads();   // protect Kc/Vt for next chunk's staging
  }

  // --- normalize and write ctx as bf16 in [B,S,H,DK] ---
#pragma unroll
  for (int nt = 0; nt < 4; ++nt) {
    const int d = nt * 16 + l15;
#pragma unroll
    for (int r = 0; r < 8; ++r) {
      const int qrow = qBase + r + hi * 8;
      const float val = ctx[nt][r] / lrow[r];
      Ctx[(((size_t)b * SEQ + qrow) * HEADS + h) * DK + d] = (bf16)val;
    }
  }
}

// ---------------------------------------------------------------------------
// Pass 3: out[m,n] = f32( Ctx[m,:] @ Wo[:,n] + bo[n] ), Ctx bf16 [4096x1024].
// Same 128x128 tiling as pass 1; A tile staged with async b128 copies.
// ---------------------------------------------------------------------------
__global__ __launch_bounds__(256)
void out_proj_kernel(const bf16* __restrict__ X, const float* __restrict__ W,
                     const float* __restrict__ bias, float* __restrict__ Out) {
  __shared__ bf16 As[128][32];
  __shared__ bf16 Wt[128][32];

  const int tid  = threadIdx.x;
  const int lane = tid & 31;
  const int wave = tid >> 5;
  const int hi   = lane >> 4;
  const int l15  = lane & 15;
  const int mBase = blockIdx.x * 128;
  const int nBase = blockIdx.y * 128;
  const int wm = wave & 3;
  const int wn = wave >> 2;

  v8f acc[2][4] = {};

  for (int kt = 0; kt < EMBED / 32; ++kt) {
    // --- stage A (already bf16): two async b128 copies per thread ---
    {
      const int r  = tid >> 1;
      const int c0 = (tid & 1) * 16;
      const bf16* src = X + (size_t)(mBase + r) * EMBED + kt * 32 + c0;
      async_copy16(src,     &As[r][c0]);
      async_copy16(src + 8, &As[r][c0 + 8]);
    }
    // --- stage W transposed: 32x128 f32 -> Wt[n][k] bf16 (16 per thread) ---
    {
      const int k  = tid & 31;
      const int n0 = (tid >> 5) * 16;
      const float* src = W + (size_t)(kt * 32 + k) * EMBED + nBase + n0;
      if (kt + 1 < EMBED / 32) __builtin_prefetch(src + 32 * EMBED, 0, 1);
#pragma unroll
      for (int i = 0; i < 16; i += 4) {
        v4f v = *(const v4f*)(src + i);
        Wt[n0 + i + 0][k] = (bf16)v[0];
        Wt[n0 + i + 1][k] = (bf16)v[1];
        Wt[n0 + i + 2][k] = (bf16)v[2];
        Wt[n0 + i + 3][k] = (bf16)v[3];
      }
    }
    wait_async0();
    __syncthreads();

    v16bf a[2], b[4];
#pragma unroll
    for (int mt = 0; mt < 2; ++mt) {
      const bf16* row = &As[wm * 32 + mt * 16 + l15][0];
      a[mt] = load_frag_pair(row + hi * 8, row + 16 + hi * 8);
    }
#pragma unroll
    for (int nt = 0; nt < 4; ++nt) {
      const bf16* col = &Wt[wn * 64 + nt * 16 + l15][0];
      b[nt] = load_frag_pair(col + hi * 8, col + 16 + hi * 8);
    }
#pragma unroll
    for (int mt = 0; mt < 2; ++mt)
#pragma unroll
      for (int nt = 0; nt < 4; ++nt)
        acc[mt][nt] = __builtin_amdgcn_wmma_f32_16x16x32_bf16(
            false, a[mt], false, b[nt], (short)0, acc[mt][nt], false, false);
    __syncthreads();
  }

#pragma unroll
  for (int mt = 0; mt < 2; ++mt) {
#pragma unroll
    for (int nt = 0; nt < 4; ++nt) {
      const int ncol = nBase + wn * 64 + nt * 16 + l15;
      const float bv = bias[ncol];
#pragma unroll
      for (int r = 0; r < 8; ++r) {
        const int mrow = mBase + wm * 32 + mt * 16 + r + hi * 8;
        Out[(size_t)mrow * EMBED + ncol] = acc[mt][nt][r] + bv;
      }
    }
  }
}

// ---------------------------------------------------------------------------
extern "C" void kernel_launch(void* const* d_in, const int* in_sizes, int n_in,
                              void* d_out, int out_size, void* d_ws, size_t ws_size,
                              hipStream_t stream) {
  const float* query = (const float*)d_in[0];
  const float* key   = (const float*)d_in[1];
  const float* value = (const float*)d_in[2];
  const int*   mask  = (const int*)d_in[3];
  const float* wq = (const float*)d_in[4];
  const float* bq = (const float*)d_in[5];
  const float* wk = (const float*)d_in[6];
  const float* bk = (const float*)d_in[7];
  const float* wv = (const float*)d_in[8];
  const float* bv = (const float*)d_in[9];
  const float* wo = (const float*)d_in[10];
  const float* bo = (const float*)d_in[11];

  const size_t elems = (size_t)BATCH * SEQ * EMBED;  // 4M elements
  bf16* Qw = (bf16*)d_ws;
  bf16* Kw = Qw + elems;
  bf16* Vw = Kw + elems;
  bf16* Cw = Vw + elems;

  dim3 blk(256);
  dim3 gProj(BATCH * SEQ / 128, EMBED / 128);  // 32 x 8
  proj_kernel<<<gProj, blk, 0, stream>>>(query, wq, bq, Qw);
  proj_kernel<<<gProj, blk, 0, stream>>>(key,   wk, bk, Kw);
  proj_kernel<<<gProj, blk, 0, stream>>>(value, wv, bv, Vw);

  dim3 gAttn(SEQ / 128, BATCH * HEADS);        // 16 x 32
  attn_kernel<<<gAttn, blk, 0, stream>>>(Qw, Kw, Vw, mask, Cw);

  out_proj_kernel<<<gProj, blk, 0, stream>>>(Cw, wo, bo, (float*)d_out);
}
